// staGCNN_9328668967085
// MI455X (gfx1250) — matrix operand order repaired and checked by reading, not compiled, not run on hardware
//
#include <hip/hip_runtime.h>

#define Nn    30000
#define Ee    240000
#define Gg    64
#define Ff    1024
#define OUTD  2048
#define INNER 512

// ---------------------------------------------------------------- WMMA GEMM
typedef __attribute__((ext_vector_type(16))) __bf16        v16bf;
typedef __attribute__((ext_vector_type(8)))  float         v8f;
typedef __attribute__((ext_vector_type(4)))  unsigned int  u32x4;

union FragBF { v16bf v; u32x4 q[2]; };
union BPack  { u32x4 q; unsigned short s[8]; };

#define BM 128
#define BN 64
#define BK 32
#define LDS_K (BK + 8)   // padded row: 40 ushorts = 80B, keeps 16B alignment

// C[M,Nc] = A[M,K](bf16) @ B[K,Nc](bf16), fp32 accumulate.
// Requires: K % 32 == 0, Nc % 64 == 0. M arbitrary (guarded).
__global__ __launch_bounds__(256) void gemm_bf16_wmma(
    const unsigned short* __restrict__ A,
    const unsigned short* __restrict__ B,
    float* __restrict__ C,
    int M, int Nc, int K)
{
  __shared__ __align__(16) unsigned short As[BM][LDS_K];
  __shared__ __align__(16) unsigned short Bt[BN][LDS_K];   // transposed: [n][k]

  const int tid  = threadIdx.x;
  const int lane = tid & 31;
  const int wave = tid >> 5;       // 8 waves
  const int wm   = wave >> 1;      // 0..3 (32 rows each)
  const int wn   = wave & 1;       // 0..1 (32 cols each)
  const int lh   = lane >> 4;      // lane half (wave32 halves)
  const int l15  = lane & 15;

  const int blockM = blockIdx.y * BM;
  const int blockN = blockIdx.x * BN;

  const v8f vzero = {0.f,0.f,0.f,0.f,0.f,0.f,0.f,0.f};
  v8f acc[2][2];
  acc[0][0]=vzero; acc[0][1]=vzero; acc[1][0]=vzero; acc[1][1]=vzero;

  const int arow = tid >> 1;          // 0..127
  const int acol = (tid & 1) * 16;    // 0 or 16
  const int brow = tid >> 3;          // 0..31
  const int bcol = (tid & 7) * 8;     // 0..56
  const u32x4 qz = {0u,0u,0u,0u};

  for (int k0 = 0; k0 < K; k0 += BK) {
    // ---- stage global -> regs
    u32x4 a0 = qz, a1 = qz;
    const int gr = blockM + arow;
    if (gr < M) {
      const u32x4* ap = (const u32x4*)(A + (size_t)gr * K + k0 + acol);
      a0 = ap[0]; a1 = ap[1];
    }
    BPack bb;
    bb.q = *(const u32x4*)(B + (size_t)(k0 + brow) * Nc + blockN + bcol);

    // ---- regs -> LDS (B stored transposed so fragment reads are contiguous)
    __syncthreads();
    *(u32x4*)&As[arow][acol]     = a0;
    *(u32x4*)&As[arow][acol + 8] = a1;
    #pragma unroll
    for (int i = 0; i < 8; ++i) Bt[bcol + i][brow] = bb.s[i];
    __syncthreads();

    if (k0 + BK < K) {   // emits global_prefetch_b8 for next K tile
      const int pr = gr < M ? gr : (M - 1);
      __builtin_prefetch(A + (size_t)pr * K + k0 + BK + acol, 0, 1);
      __builtin_prefetch(B + (size_t)(k0 + BK + brow) * Nc + blockN + bcol, 0, 1);
    }

    // ---- fragments per documented 16-bit WMMA layouts
    FragBF aF[2], bF[2];
    #pragma unroll
    for (int tm = 0; tm < 2; ++tm) {
      const unsigned short* ab = &As[wm*32 + tm*16 + l15][0];
      aF[tm].q[0] = *(const u32x4*)(ab + 8*lh);        // K = 8*lh .. +7
      aF[tm].q[1] = *(const u32x4*)(ab + 16 + 8*lh);   // K = 16+8*lh .. +7
    }
    #pragma unroll
    for (int tn = 0; tn < 2; ++tn) {
      const unsigned short* bp = &Bt[wn*32 + tn*16 + l15][0];
      bF[tn].q[0] = *(const u32x4*)(bp + 8*lh);        // K = 8*lh .. +7
      bF[tn].q[1] = *(const u32x4*)(bp + 16 + 8*lh);   // K = 16+8*lh .. +7
    }
    #pragma unroll
    for (int tm = 0; tm < 2; ++tm)
      #pragma unroll
      for (int tn = 0; tn < 2; ++tn)
        acc[tm][tn] = __builtin_amdgcn_wmma_f32_16x16x32_bf16(
            false, aF[tm].v, false, bF[tn].v, (short)0, acc[tm][tn], false, false);
  }

  // ---- store: VGPR r -> M = r + 8*lh, N = l15
  #pragma unroll
  for (int tm = 0; tm < 2; ++tm) {
    const int row0 = blockM + wm*32 + tm*16 + lh*8;
    #pragma unroll
    for (int tn = 0; tn < 2; ++tn) {
      const int col = blockN + wn*32 + tn*16 + l15;
      #pragma unroll
      for (int r = 0; r < 8; ++r) {
        const int row = row0 + r;
        if (row < M) C[(size_t)row * Nc + col] = acc[tm][tn][r];
      }
    }
  }
}

// ---------------------------------------------------------------- helpers
__device__ __forceinline__ unsigned short f32_to_bf16(float f) {
  unsigned u = __float_as_uint(f);
  u += 0x7FFFu + ((u >> 16) & 1u);      // round-to-nearest-even
  return (unsigned short)(u >> 16);
}
__device__ __forceinline__ unsigned enc_f32(float f) {   // order-preserving
  unsigned u = __float_as_uint(f);
  return (u & 0x80000000u) ? ~u : (u | 0x80000000u);
}
__device__ __forceinline__ float dec_f32(unsigned e) {
  unsigned u = (e & 0x80000000u) ? (e & 0x7FFFFFFFu) : ~e;
  return __uint_as_float(u);
}

__global__ void fill_f32(float* p, float v, int n) {
  int i = blockIdx.x * 256 + threadIdx.x; if (i < n) p[i] = v;
}
__global__ void fill_u32(unsigned* p, unsigned v, int n) {
  int i = blockIdx.x * 256 + threadIdx.x; if (i < n) p[i] = v;
}
__global__ void cvt_f32_bf16(const float* __restrict__ s, unsigned short* __restrict__ d, int n) {
  int i = blockIdx.x * 256 + threadIdx.x; if (i < n) d[i] = f32_to_bf16(s[i]);
}
__global__ void deg_accum(const int* __restrict__ dst, float* deg, int e) {
  int i = blockIdx.x * 256 + threadIdx.x; if (i < e) atomicAdd(&deg[dst[i]], 1.0f);
}
__global__ void deg_finish(const float* __restrict__ deg, float* __restrict__ dinv,
                           float* __restrict__ dinv2, int n) {
  int i = blockIdx.x * 256 + threadIdx.x;
  if (i < n) { float d = deg[i]; dinv[i] = rsqrtf(d); dinv2[i] = 1.0f / d; }
}

// out[dst] += H[src] * dinv[src]*dinv[dst]   (one block per edge, 1024 feats)
__global__ __launch_bounds__(256) void edge_scatter(
    const float* __restrict__ H, const int* __restrict__ src, const int* __restrict__ dst,
    const float* __restrict__ dinv, float* __restrict__ out)
{
  const int e = blockIdx.x;
  const int s = src[e], d = dst[e];
  const float coef = dinv[s] * dinv[d];
  const float4* hs = (const float4*)(H + (size_t)s * Ff);
  float* od = out + (size_t)d * Ff;
  const float4 v = hs[threadIdx.x];
  const int f = threadIdx.x * 4;
  atomicAdd(od + f + 0, v.x * coef);
  atomicAdd(od + f + 1, v.y * coef);
  atomicAdd(od + f + 2, v.z * coef);
  atomicAdd(od + f + 3, v.w * coef);
}

// h = x + relu(agg + H*dinv2[row] + b[col])  -> bf16
__global__ void combine_res_relu(const float* __restrict__ x, const float* __restrict__ H,
                                 const float* __restrict__ agg, const float* __restrict__ dinv2,
                                 const float* __restrict__ bias, unsigned short* __restrict__ hb, int n)
{
  int i = blockIdx.x * 256 + threadIdx.x;
  if (i < n) {
    int row = i >> 10, col = i & 1023;
    float v = agg[i] + H[i] * dinv2[row] + bias[col];
    hb[i] = f32_to_bf16(fmaxf(v, 0.f) + x[i]);
  }
}
// out = agg + H*dinv2[row] + b[col]
__global__ void combine_lin(const float* __restrict__ H, const float* __restrict__ agg,
                            const float* __restrict__ dinv2, const float* __restrict__ bias,
                            float* __restrict__ out, int n)
{
  int i = blockIdx.x * 256 + threadIdx.x;
  if (i < n) {
    int row = i >> 10, col = i & 1023;
    out[i] = agg[i] + H[i] * dinv2[row] + bias[col];
  }
}

__global__ __launch_bounds__(256) void segmax_scatter(
    const float* __restrict__ h2, const int* __restrict__ batch, unsigned* __restrict__ enc)
{
  const int nIdx = blockIdx.x;
  const int g = batch[nIdx];
  const float* row = h2 + (size_t)nIdx * Ff;
  unsigned* eg = enc + (size_t)g * Ff;
  for (int f = threadIdx.x; f < Ff; f += 256) atomicMax(eg + f, enc_f32(row[f]));
}
__global__ void segmax_finish(const unsigned* __restrict__ enc, unsigned short* __restrict__ pb, int n) {
  int i = blockIdx.x * 256 + threadIdx.x; if (i < n) pb[i] = f32_to_bf16(dec_f32(enc[i]));
}

// relu(in + bias) -> bf16 written into concat buffer at column offset
__global__ void bias_relu_xc(const float* __restrict__ in, const float* __restrict__ bias,
                             unsigned short* __restrict__ xc, int cols, int ld, int coloff, int n)
{
  int i = blockIdx.x * 256 + threadIdx.x;
  if (i < n) {
    int g = i / cols, j = i - g * cols;
    xc[(size_t)g * ld + coloff + j] = f32_to_bf16(fmaxf(in[i] + bias[j], 0.f));
  }
}
__global__ void bias_relu_f32(const float* __restrict__ in, const float* __restrict__ bias,
                              float* __restrict__ out, int cols, int n)
{
  int i = blockIdx.x * 256 + threadIdx.x;
  if (i < n) { int g = i / cols, j = i - g * cols; out[i] = fmaxf(in[i] + bias[j], 0.f); }
}

__global__ __launch_bounds__(256) void mlp2_kernel(const float* __restrict__ hid,
                                                   const float* __restrict__ W2,
                                                   const float* __restrict__ b2,
                                                   float* __restrict__ out)
{
  __shared__ float red[256];
  const int g = blockIdx.x;
  float p = 0.f;
  for (int k = threadIdx.x; k < INNER; k += 256) p += hid[(size_t)g * INNER + k] * W2[k];
  red[threadIdx.x] = p; __syncthreads();
  for (int s = 128; s > 0; s >>= 1) {
    if (threadIdx.x < s) red[threadIdx.x] += red[threadIdx.x + s];
    __syncthreads();
  }
  if (threadIdx.x == 0) out[g] = red[0] + b2[0];
}

// ---------------------------------------------------------------- host
static inline int cdiv(int a, int b) { return (a + b - 1) / b; }

extern "C" void kernel_launch(void* const* d_in, const int* in_sizes, int n_in,
                              void* d_out, int out_size, void* d_ws, size_t ws_size,
                              hipStream_t stream)
{
  (void)in_sizes; (void)n_in; (void)out_size;
  const float* wild_x  = (const float*)d_in[0];
  const float* mut_x   = (const float*)d_in[1];
  const int*   wild_ei = (const int*)d_in[2];
  const int*   mut_ei  = (const int*)d_in[3];
  const int*   wild_b  = (const int*)d_in[4];
  const int*   mut_b   = (const int*)d_in[5];
  const float* wW1 = (const float*)d_in[6];  const float* wb1 = (const float*)d_in[7];
  const float* wW2 = (const float*)d_in[8];  const float* wb2 = (const float*)d_in[9];
  const float* mW1 = (const float*)d_in[10]; const float* mb1 = (const float*)d_in[11];
  const float* mW2 = (const float*)d_in[12]; const float* mb2 = (const float*)d_in[13];
  const float* wFc = (const float*)d_in[14]; const float* wFb = (const float*)d_in[15];
  const float* mFc = (const float*)d_in[16]; const float* mFb = (const float*)d_in[17];
  const float* mlpW1 = (const float*)d_in[18]; const float* mlpB1 = (const float*)d_in[19];
  const float* mlpW2 = (const float*)d_in[20]; const float* mlpB2 = (const float*)d_in[21];

  // workspace layout
  char* base = (char*)d_ws;
  size_t off = 0;
  auto take = [&](size_t bytes) { void* p = base + off; off = (off + bytes + 255) & ~(size_t)255; return p; };
  const size_t NF = (size_t)Nn * Ff;
  unsigned short* XB   = (unsigned short*)take(NF * 2);          // x / h in bf16
  float*          HF   = (float*)take(NF * 4);                   // GEMM out
  float*          AG   = (float*)take(NF * 4);                   // aggregation
  float*          DEG  = (float*)take((size_t)Nn * 4);
  float*          DINV = (float*)take((size_t)Nn * 4);
  float*          DIN2 = (float*)take((size_t)Nn * 4);
  unsigned short* WB   = (unsigned short*)take((size_t)2 * OUTD * INNER * 2); // largest weight (2M elems)
  unsigned*       ENC  = (unsigned*)take((size_t)Gg * Ff * 4);
  unsigned short* PB   = (unsigned short*)take((size_t)Gg * Ff * 2);
  float*          FCO  = (float*)take((size_t)Gg * OUTD * 4);
  unsigned short* XC   = (unsigned short*)take((size_t)Gg * 2 * OUTD * 2);
  float*          HID  = (float*)take((size_t)Gg * INNER * 4);
  if (off > ws_size) return;

  const int nf  = (int)NF;
  const int gNF = cdiv(nf, 256);

  auto run_branch = [&](const float* x, const int* ei, const int* batch,
                        const float* W1, const float* b1c,
                        const float* W2, const float* b2c,
                        const float* Wfc, const float* bfc, int xc_off)
  {
    const int* src = ei;
    const int* dst = ei + Ee;
    // degree normalization (deg = in-degree + 1)
    fill_f32<<<cdiv(Nn,256), 256, 0, stream>>>(DEG, 1.0f, Nn);
    deg_accum<<<cdiv(Ee,256), 256, 0, stream>>>(dst, DEG, Ee);
    deg_finish<<<cdiv(Nn,256), 256, 0, stream>>>(DEG, DINV, DIN2, Nn);
    // conv1: H = x @ W1
    cvt_f32_bf16<<<gNF, 256, 0, stream>>>(x, XB, nf);
    cvt_f32_bf16<<<cdiv(Ff*Ff,256), 256, 0, stream>>>(W1, WB, Ff*Ff);
    gemm_bf16_wmma<<<dim3(Ff/BN, cdiv(Nn,BM)), 256, 0, stream>>>(XB, WB, HF, Nn, Ff, Ff);
    fill_f32<<<gNF, 256, 0, stream>>>(AG, 0.0f, nf);
    edge_scatter<<<Ee, 256, 0, stream>>>(HF, src, dst, DINV, AG);
    combine_res_relu<<<gNF, 256, 0, stream>>>(x, HF, AG, DIN2, b1c, XB, nf); // XB <- h (bf16)
    // conv2: H = h @ W2
    cvt_f32_bf16<<<cdiv(Ff*Ff,256), 256, 0, stream>>>(W2, WB, Ff*Ff);
    gemm_bf16_wmma<<<dim3(Ff/BN, cdiv(Nn,BM)), 256, 0, stream>>>(XB, WB, HF, Nn, Ff, Ff);
    fill_f32<<<gNF, 256, 0, stream>>>(AG, 0.0f, nf);
    edge_scatter<<<Ee, 256, 0, stream>>>(HF, src, dst, DINV, AG);
    combine_lin<<<gNF, 256, 0, stream>>>(HF, AG, DIN2, b2c, AG, nf);        // AG <- h2 (f32)
    // segment max pool -> bf16 [G, F]
    fill_u32<<<cdiv(Gg*Ff,256), 256, 0, stream>>>(ENC, 0u, Gg*Ff);
    segmax_scatter<<<Nn, 256, 0, stream>>>(AG, batch, ENC);
    segmax_finish<<<cdiv(Gg*Ff,256), 256, 0, stream>>>(ENC, PB, Gg*Ff);
    // fc1 + relu into concat buffer half
    cvt_f32_bf16<<<cdiv(Ff*OUTD,256), 256, 0, stream>>>(Wfc, WB, Ff*OUTD);
    gemm_bf16_wmma<<<dim3(OUTD/BN, cdiv(Gg,BM)), 256, 0, stream>>>(PB, WB, FCO, Gg, OUTD, Ff);
    bias_relu_xc<<<cdiv(Gg*OUTD,256), 256, 0, stream>>>(FCO, bfc, XC, OUTD, 2*OUTD, xc_off, Gg*OUTD);
  };

  // xc = concat([mutant, wild], axis=1)
  run_branch(mut_x,  mut_ei,  mut_b,  mW1, mb1, mW2, mb2, mFc, mFb, 0);
  run_branch(wild_x, wild_ei, wild_b, wW1, wb1, wW2, wb2, wFc, wFb, OUTD);

  // MLP: hid = relu(xc @ W1 + b1); out = hid @ W2 + b2
  cvt_f32_bf16<<<cdiv(2*OUTD*INNER,256), 256, 0, stream>>>(mlpW1, WB, 2*OUTD*INNER);
  gemm_bf16_wmma<<<dim3(INNER/BN, cdiv(Gg,BM)), 256, 0, stream>>>(XC, WB, FCO, Gg, INNER, 2*OUTD);
  bias_relu_f32<<<cdiv(Gg*INNER,256), 256, 0, stream>>>(FCO, mlpB1, HID, INNER, Gg*INNER);
  mlp2_kernel<<<Gg, 256, 0, stream>>>(HID, mlpW2, mlpB2, (float*)d_out);
}
